// LSTMEncoder_5377299055298
// MI455X (gfx1250) — matrix operand order repaired and checked by reading, not compiled, not run on hardware
//
#include <hip/hip_runtime.h>

// ---------------- problem constants ----------------
#define Bn 32
#define Tn 512
#define Dn 512
#define Hn 512
#define Ln 2
#define GHn 2048        // 4*H
#define NWG 16          // workgroups (column slices of H)
#define COLS 128        // gate columns per WG (4 gates x 32)
#define HSL 32          // H columns per WG
#define NTHREADS 256    // 8 wave32

typedef __attribute__((ext_vector_type(16))) __bf16 v16bf;
typedef __attribute__((ext_vector_type(8)))  __bf16 v8bf;
typedef __attribute__((ext_vector_type(8)))  float  v8f;

// LDS carve-up (dynamic shared): 128KB W slice + 32KB A tile + 16KB gate buf + 4KB cell
#define LDS_W_OFF 0
#define LDS_A_OFF (COLS * 512 * 2)                  // 131072
#define LDS_G_OFF (LDS_A_OFF + 32 * 512 * 2)        // +32768
#define LDS_C_OFF (LDS_G_OFF + COLS * 32 * 4)       // +16384
#define SMEM_BYTES (LDS_C_OFF + Bn * HSL * 4)       // 184320

__device__ __forceinline__ __bf16 f2bf(float f) {
  union { float f; unsigned u; } a; a.f = f;
  unsigned u = a.u + 0x7FFFu + ((a.u >> 16) & 1u);   // round-to-nearest-even
  union { unsigned short s; __bf16 b; } o; o.s = (unsigned short)(u >> 16);
  return o.b;
}
__device__ __forceinline__ float bf2f(__bf16 b) {
  union { unsigned short s; __bf16 b; } i; i.b = b;
  union { unsigned u; float f; } o; o.u = ((unsigned)i.s) << 16;
  return o.f;
}
__device__ __forceinline__ float sigm(float x) { return 1.0f / (1.0f + __expf(-x)); }

// ---- CDNA5 async global->LDS copy (ASYNCcnt), GVS mode: addr = SADDR64 + VADDR32 ----
// VDST VGPR = byte offset within the wave's LDS allocation (dynamic smem starts at 0).
__device__ __forceinline__ void async_g2l_b128(unsigned lds_byte_off,
                                               unsigned long long gbase,
                                               unsigned gbyte_off) {
  asm volatile("global_load_async_to_lds_b128 %0, %1, %2"
               :
               : "v"(lds_byte_off), "v"(gbyte_off), "s"(gbase)
               : "memory");
}
__device__ __forceinline__ void wait_async0() {
  asm volatile("s_wait_asynccnt 0" ::: "memory");
}

// A-matrix 16x32 bf16 fragment: lane m=lane&15, khalf=lane>>4;
// elements 0-15 hold K=khalf*16..+15 (matches B fragment)  -> two 16B LDS reads
__device__ __forceinline__ v16bf load_a_frag(const __bf16* ldsA, int m0, int k0, int lane) {
  const __bf16* p = ldsA + (m0 + (lane & 15)) * 512 + k0 + ((lane >> 4) << 4);
  v8bf lo = *(const v8bf*)p;
  v8bf hi = *(const v8bf*)(p + 8);
  v16bf r;
#pragma unroll
  for (int i = 0; i < 8; ++i) { r[i] = lo[i]; r[i + 8] = hi[i]; }
  return r;
}
// B 32x16 bf16 fragment, k order chosen to match A: lane n=lane&15; lanes 0-15 K=0-15, 16-31 K=16-31.
// Our LDS weight tile is [col][K] (i.e. B^T), so one 32B contiguous read per lane.
__device__ __forceinline__ v16bf load_b_frag(const __bf16* ldsW, int row, int k0, int lane) {
  const __bf16* p = ldsW + row * 512 + k0 + ((lane >> 4) << 4);
  return *(const v16bf*)p;
}

__device__ __forceinline__ void grid_sync(unsigned* cnt, unsigned goal) {
  __syncthreads();
  if (threadIdx.x == 0) {
    __threadfence();                       // release
    atomicAdd(cnt, 1u);
    while (__hip_atomic_load(cnt, __ATOMIC_RELAXED, __HIP_MEMORY_SCOPE_AGENT) < goal)
      __builtin_amdgcn_s_sleep(2);
  }
  __syncthreads();
  __threadfence();                         // acquire
}

__device__ __forceinline__ void gemm_tile(const __bf16* ldsA, const __bf16* ldsW,
                                          int m0, int cl0, int cl1, int lane,
                                          v8f& acc0, v8f& acc1) {
#pragma unroll 4
  for (int k0 = 0; k0 < 512; k0 += 32) {
    v16bf af = load_a_frag(ldsA, m0, k0, lane);
    v16bf b0 = load_b_frag(ldsW, cl0, k0, lane);
    v16bf b1 = load_b_frag(ldsW, cl1, k0, lane);
    acc0 = __builtin_amdgcn_wmma_f32_16x16x32_bf16(false, af, false, b0, (short)0, acc0, false, false);
    acc1 = __builtin_amdgcn_wmma_f32_16x16x32_bf16(false, af, false, b1, (short)0, acc1, false, false);
  }
}

extern "C" __global__ void __launch_bounds__(NTHREADS, 1)
lstm2_persist(const float* __restrict__ x,    // [B,T,D]
              const float* __restrict__ Wih,  // [L,4H,D]
              const float* __restrict__ Whh,  // [L,4H,H]
              const float* __restrict__ bih,  // [L,4H]
              const float* __restrict__ bhh,  // [L,4H]
              float* __restrict__ out,        // h_n [L,B,H] ++ c_n [L,B,H]
              unsigned* __restrict__ cnt,     // grid barrier counter (zeroed)
              __bf16* __restrict__ hbuf0,     // [B,H] (zeroed)
              __bf16* __restrict__ hbuf1,     // [B,H] (zeroed)
              __bf16* __restrict__ ybuf,      // [B,T,H] layer-0 outputs
              __bf16* __restrict__ gxbuf)     // [T][NWG][B][COLS]
{
  extern __shared__ char smem[];
  __bf16* ldsW = (__bf16*)(smem + LDS_W_OFF);   // [COLS][512] weight slice (bf16)
  __bf16* ldsA = (__bf16*)(smem + LDS_A_OFF);   // [32][512] activation tile (bf16)
  float*  gbuf = (float*)(smem + LDS_G_OFF);    // [COLS][32] raw gate pre-acts
  float*  csl  = (float*)(smem + LDS_C_OFF);    // [32][32] cell state slice

  const int tid  = threadIdx.x;
  const int wg   = blockIdx.x;
  const int lane = tid & 31;
  const int wave = tid >> 5;
  const int m0   = (wave >> 2) * 16;   // row (batch) base: 0 or 16
  const int n0   = (wave & 3) * 32;    // local column base: wave owns one gate slice
  const int cl0  = n0 + (lane & 15);   // local cols of the two accumulators
  const int cl1  = cl0 + 16;
  const int rofs = (lane >> 4) << 3;   // +8 rows for lanes 16-31 (C layout)
  unsigned phase = 0;
  __bf16* hb[2] = { hbuf0, hbuf1 };

  // per-thread tile-staging coordinates: thread covers row tb, 64 halfs at tk0
  const int tb  = tid >> 3;
  const int tk0 = (tid & 7) * 64;

  for (int layer = 0; layer < Ln; ++layer) {
    // ---- stage W_ih column-slice into LDS (fp32 -> bf16), [local_col][K] ----
    const float* WihL = Wih + (size_t)layer * GHn * Dn;
    for (int idx = tid; idx < COLS * 512; idx += NTHREADS) {
      int r = idx >> 9, k = idx & 511;
      int g = (r >> 5) * Hn + wg * HSL + (r & 31);     // gate-major global row
      ldsW[idx] = f2bf(WihL[(size_t)g * Dn + k]);
    }
    float bias0, bias1;
    {
      int g0 = (cl0 >> 5) * Hn + wg * HSL + (cl0 & 31);
      int g1 = (cl1 >> 5) * Hn + wg * HSL + (cl1 & 31);
      bias0 = bih[layer * GHn + g0] + bhh[layer * GHn + g0];
      bias1 = bih[layer * GHn + g1] + bhh[layer * GHn + g1];
    }
    __syncthreads();

    // ---- Phase 1: gx[t][wg][b][col] = x_l[b,t,:] @ Wih_slice^T + (bih+bhh) ----
    for (int t = 0; t < Tn; ++t) {
      if (layer == 0) {
        // f32 source needs conversion: manual staging
        const float* src = x + ((size_t)tb * Tn + t) * Dn + tk0;
#pragma unroll
        for (int i = 0; i < 64; ++i) ldsA[tb * 512 + tk0 + i] = f2bf(src[i]);
      } else {
        // bf16 source: async DMA straight into LDS (no VGPR staging), 8x 16B per thread
        unsigned lbase = LDS_A_OFF + (unsigned)(tb * 512 + tk0) * 2u;
        unsigned gbase_off = (unsigned)((((size_t)tb * Tn + t) * Hn + tk0) * 2u);
#pragma unroll
        for (int i = 0; i < 8; ++i)
          async_g2l_b128(lbase + i * 16u, (unsigned long long)(uintptr_t)ybuf,
                         gbase_off + i * 16u);
        wait_async0();
      }
      __syncthreads();
      v8f acc0 = {0.f,0.f,0.f,0.f,0.f,0.f,0.f,0.f};
      v8f acc1 = {0.f,0.f,0.f,0.f,0.f,0.f,0.f,0.f};
      gemm_tile(ldsA, ldsW, m0, cl0, cl1, lane, acc0, acc1);
      size_t base = ((size_t)t * NWG + wg) * Bn * COLS;
#pragma unroll
      for (int j = 0; j < 8; ++j) {
        int b = m0 + rofs + j;                       // C layout: M = j (+8 upper lanes)
        gxbuf[base + (size_t)b * COLS + cl0] = f2bf(acc0[j] + bias0);
        gxbuf[base + (size_t)b * COLS + cl1] = f2bf(acc1[j] + bias1);
      }
      __syncthreads();
    }

    // ---- stage W_hh slice; reset cell slice; (layer>0) re-zero own h slices ----
    const float* WhhL = Whh + (size_t)layer * GHn * Hn;
    for (int idx = tid; idx < COLS * 512; idx += NTHREADS) {
      int r = idx >> 9, k = idx & 511;
      int g = (r >> 5) * Hn + wg * HSL + (r & 31);
      ldsW[idx] = f2bf(WhhL[(size_t)g * Hn + k]);
    }
    for (int idx = tid; idx < Bn * HSL; idx += NTHREADS) csl[idx] = 0.f;
    if (layer > 0) {
      __bf16 z = f2bf(0.f);
      for (int idx = tid; idx < Bn * HSL; idx += NTHREADS) {
        int b = idx >> 5, h = idx & 31;
        hbuf0[(size_t)b * Hn + wg * HSL + h] = z;
        hbuf1[(size_t)b * Hn + wg * HSL + h] = z;
      }
    }
    grid_sync(cnt, ++phase * NWG);

    // ---- Phase 2: 512-step recurrence, Whh slice LDS-resident ----
    for (int t = 0; t < Tn; ++t) {
      const __bf16* hcur = hb[t & 1];
      __bf16* hnxt = hb[(t + 1) & 1];
      {
        // async DMA the full h vector [32][512] bf16 into LDS
        unsigned lbase = LDS_A_OFF + (unsigned)(tb * 512 + tk0) * 2u;
        unsigned gbase_off = (unsigned)(((size_t)tb * Hn + tk0) * 2u);
#pragma unroll
        for (int i = 0; i < 8; ++i)
          async_g2l_b128(lbase + i * 16u, (unsigned long long)(uintptr_t)hcur,
                         gbase_off + i * 16u);
        wait_async0();
      }
      __syncthreads();
      v8f acc0 = {0.f,0.f,0.f,0.f,0.f,0.f,0.f,0.f};
      v8f acc1 = {0.f,0.f,0.f,0.f,0.f,0.f,0.f,0.f};
      gemm_tile(ldsA, ldsW, m0, cl0, cl1, lane, acc0, acc1);
#pragma unroll
      for (int j = 0; j < 8; ++j) {
        int b = m0 + rofs + j;
        gbuf[cl0 * 32 + b] = acc0[j];
        gbuf[cl1 * 32 + b] = acc1[j];
      }
      __syncthreads();
      {
        int b = tb, hbase = (tid & 7) * 4;
        size_t gxb = (((size_t)t * NWG + wg) * Bn + b) * COLS;
        if (t + 1 < Tn)  // warm L2 for next step's gx slab (global_prefetch_b8)
          __builtin_prefetch(gxbuf + (((size_t)(t + 1) * NWG + wg) * Bn + b) * COLS, 0, 1);
#pragma unroll
        for (int u = 0; u < 4; ++u) {
          int h = hbase + u;
          float gi = gbuf[(0 * 32 + h) * 32 + b] + bf2f(gxbuf[gxb + 0  + h]);
          float gf = gbuf[(1 * 32 + h) * 32 + b] + bf2f(gxbuf[gxb + 32 + h]);
          float gg = gbuf[(2 * 32 + h) * 32 + b] + bf2f(gxbuf[gxb + 64 + h]);
          float go = gbuf[(3 * 32 + h) * 32 + b] + bf2f(gxbuf[gxb + 96 + h]);
          float iv = sigm(gi), fv = sigm(gf), gv = tanhf(gg), ov = sigm(go);
          float c  = fv * csl[b * 32 + h] + iv * gv;
          csl[b * 32 + h] = c;
          float hv = ov * tanhf(c);
          int colg = wg * HSL + h;
          hnxt[(size_t)b * Hn + colg] = f2bf(hv);
          if (layer == 0) ybuf[((size_t)b * Tn + t) * Hn + colg] = f2bf(hv);
          if (t == Tn - 1) {
            out[((size_t)layer * Bn + b) * Hn + colg] = hv;                       // h_n
            out[(size_t)Ln * Bn * Hn + ((size_t)layer * Bn + b) * Hn + colg] = c; // c_n
          }
        }
      }
      grid_sync(cnt, ++phase * NWG);
    }
  }
}

extern "C" void kernel_launch(void* const* d_in, const int* in_sizes, int n_in,
                              void* d_out, int out_size, void* d_ws, size_t ws_size,
                              hipStream_t stream) {
  const float* x   = (const float*)d_in[0];
  const float* Wih = (const float*)d_in[1];
  const float* Whh = (const float*)d_in[2];
  const float* bih = (const float*)d_in[3];
  const float* bhh = (const float*)d_in[4];
  float* out = (float*)d_out;

  char* ws = (char*)d_ws;
  unsigned* cnt = (unsigned*)ws;                                  // 256 B
  __bf16* hbuf0 = (__bf16*)(ws + 256);                            // 32 KB
  __bf16* hbuf1 = (__bf16*)(ws + 256 + 32768);                    // 32 KB
  __bf16* ybuf  = (__bf16*)(ws + 256 + 65536);                    // 16 MB
  __bf16* gxbuf = (__bf16*)(ws + 256 + 65536 + (size_t)Bn * Tn * Hn * 2); // 64 MB

  // capture-safe zeroing of barrier counter + both h buffers each launch
  hipMemsetAsync(ws, 0, 256 + 65536, stream);

  hipFuncSetAttribute((const void*)lstm2_persist,
                      hipFuncAttributeMaxDynamicSharedMemorySize, SMEM_BYTES);
  lstm2_persist<<<dim3(NWG), dim3(NTHREADS), SMEM_BYTES, stream>>>(
      x, Wih, Whh, bih, bhh, out, cnt, hbuf0, hbuf1, ybuf, gxbuf);
}
